// CombinedGraphLayer_19207093748410
// MI455X (gfx1250) — matrix operand order repaired and checked by reading, not compiled, not run on hardware
//
#include <hip/hip_runtime.h>
#include <hip/hip_bf16.h>
#include <math.h>

// ---------------- problem constants ----------------
constexpr int kB    = 8;
constexpr int kN    = 12800;
constexpr int kF    = 256;
constexpr int kDD   = 128;   // DIST_DIM
constexpr int kFH   = 128;   // FFN_HID
constexpr int kOUT  = 256;
constexpr int kBS   = 256;   // bin size
constexpr int kNCH  = kN / kBS;       // 50 chunks per batch
constexpr int kNT1  = kN / 128;       // 100 tiles of 128 tokens (kernel 1)
constexpr int kNROT = 25;             // N_BINS // 2
constexpr int kNBINS = 50;
constexpr int kNBKEY = 100;           // bin_idx can reach 49+49=98 when masked
constexpr int kROTLD = 100;           // rotations leading dim (MAX_BINS//2)

typedef __bf16 bf16_t;
typedef __attribute__((ext_vector_type(16))) __bf16 v16bf;
typedef __attribute__((ext_vector_type(8)))  float  v8f;
typedef __attribute__((ext_vector_type(4)))  int    v4i;

// ---------------- async global->LDS (CDNA5 ASYNCcnt path), guarded ----------------
#if defined(__HIP_DEVICE_COMPILE__) && \
    __has_builtin(__builtin_amdgcn_global_load_async_to_lds_b128) && \
    __has_builtin(__builtin_amdgcn_s_wait_asynccnt)
#define ASYNC_LDS 1
typedef __attribute__((address_space(1))) v4i as1_v4i;
typedef __attribute__((address_space(3))) v4i as3_v4i;
__device__ __forceinline__ void async_copy_b128(const void* g, void* l) {
  __builtin_amdgcn_global_load_async_to_lds_b128((as1_v4i*)(void*)g, (as3_v4i*)l, 0, 0);
}
__device__ __forceinline__ void async_wait0() { __builtin_amdgcn_s_wait_asynccnt(0); }
#else
#define ASYNC_LDS 0
#endif

// ---------------- WMMA helpers (CDNA5 wave32 layouts, CDNA5 ISA) ----------------
__device__ __forceinline__ v8f wmma_bf16(v16bf a, v16bf b, v8f c) {
  // D = A(16x32 bf16) * B(32x16 bf16) + C(16x16 f32)
  return __builtin_amdgcn_wmma_f32_16x16x32_bf16(
      /*neg_a=*/false, a, /*neg_b=*/false, b,
      /*c_mod=*/(short)0, c, /*reuse_a=*/false, /*reuse_b=*/false);
}

// A fragment: 16x32 (MxK) bf16, row-major source with leading dim ld.
// lanes 0-15: M=0..15, K = 0..15; lanes 16-31: M=0..15, K = 16..31
__device__ __forceinline__ v16bf load_a_frag(const bf16_t* p, int ld) {
  const int lane = threadIdx.x & 31;
  const int m  = lane & 15;
  const int kb = (lane >> 4) << 4;          // 0 or 16
  v16bf a;
#pragma unroll
  for (int e = 0; e < 16; ++e) {
    const int k = kb + e;
    a[e] = p[m * ld + k];
  }
  return a;
}

// B fragment: 32x16 (KxN) bf16 from a row-major KxN source with leading dim ld.
// lanes 0-15: K=0..15 ; lanes 16-31: K=16..31 ; N = lane&15
__device__ __forceinline__ v16bf load_b_frag(const bf16_t* p, int ld) {
  const int lane = threadIdx.x & 31;
  const int n  = lane & 15;
  const int kb = (lane >> 4) << 4;          // 0 or 16
  v16bf b;
#pragma unroll
  for (int e = 0; e < 16; ++e) b[e] = p[(kb + e) * ld + n];
  return b;
}

// B fragment where B = X^T and X is row-major NxK (leading dim ld): B[k][n] = X[n][k]
__device__ __forceinline__ v16bf load_bt_frag(const bf16_t* p, int ld) {
  const int lane = threadIdx.x & 31;
  const int n  = lane & 15;
  const int kb = (lane >> 4) << 4;
  v16bf b;
#pragma unroll
  for (int e = 0; e < 16; ++e) b[e] = p[n * ld + kb + e];
  return b;
}

__device__ __forceinline__ float eluf(float v) { return v > 0.f ? v : expf(v) - 1.f; }

// Block GEMM: C(MxNout,f32,LDS) = A(MxK,bf16) * W(KxNout,bf16). 256 threads = 8 waves.
template <int M, int K, int Nout>
__device__ __forceinline__ void gemm_block(const bf16_t* A, const bf16_t* W, float* C) {
  const int wave = threadIdx.x >> 5;
  const int lane = threadIdx.x & 31;
  constexpr int MT = M / 16, NT = Nout / 16;
  for (int t = wave; t < MT * NT; t += 8) {
    const int ti = t / NT, tj = t % NT;
    v8f acc = {};
#pragma unroll
    for (int k0 = 0; k0 < K; k0 += 32) {
      __builtin_prefetch(W + (k0 + 32) * Nout + tj * 16, 0, 3);
      v16bf a = load_a_frag(A + ti * 16 * K + k0, K);
      v16bf b = load_b_frag(W + k0 * Nout + tj * 16, Nout);
      acc = wmma_bf16(a, b, acc);
    }
    const int col = tj * 16 + (lane & 15);
    const int rb  = ti * 16 + ((lane >> 4) << 3);
#pragma unroll
    for (int q = 0; q < 8; ++q) C[(rb + q) * Nout + col] = acc[q];
  }
}

// ---------------- k0: fp32 -> bf16 weight conversion ----------------
__global__ void k0_cvt(const float* __restrict__ src, bf16_t* __restrict__ dst, int n) {
  int i = blockIdx.x * 256 + threadIdx.x;
  if (i < n) dst[i] = (bf16_t)src[i];
}

// ---------------- k1: LN1 + FFN (3 WMMA GEMMs) + LSH bin assignment ----------------
__global__ __launch_bounds__(256) void k1_ln_ffn_lsh(
    const float* __restrict__ x, const unsigned char* __restrict__ msk,
    const float* ln1_g, const float* ln1_b,
    const float* l0g, const float* l0b, const bf16_t* w0, const float* b0,
    const float* l1g, const float* l1b, const bf16_t* w1, const float* b1,
    const float* l2g, const float* l2b, const bf16_t* w2, const float* b2,
    const float* __restrict__ rot,
    bf16_t* __restrict__ xn_bf, bf16_t* __restrict__ xd_bf, int* __restrict__ bin_idx) {
  extern __shared__ char smem[];
  bf16_t* Abuf = (bf16_t*)smem;                       // 128x256 bf16 (64KB), reused 128x128
  float*  Cbuf = (float*)(smem + 128 * 256 * 2);      // 128x128 f32 (64KB)

  const int b  = blockIdx.x / kNT1;
  const int t0 = (blockIdx.x % kNT1) * 128;
  const int r  = threadIdx.x;

  if (r < 128) {
    const int token = t0 + r;
    const size_t xb = ((size_t)b * kN + token) * kF;
    // pass A: stats of x
    float s = 0.f, ss = 0.f;
    for (int f = 0; f < kF; ++f) { float v = x[xb + f]; s += v; ss += v * v; }
    float mean = s / kF, rs = rsqrtf(ss / kF - mean * mean + 1e-6f);
    // pass B: xn = LN1(x), store bf16, gather stats of xn
    float s2 = 0.f, ss2 = 0.f;
    for (int f = 0; f < kF; ++f) {
      float v = (x[xb + f] - mean) * rs * ln1_g[f] + ln1_b[f];
      xn_bf[xb + f] = (bf16_t)v;
      s2 += v; ss2 += v * v;
    }
    float mean2 = s2 / kF, rs2 = rsqrtf(ss2 / kF - mean2 * mean2 + 1e-3f);
    // pass C: A0 = ffn_ln0(xn) into LDS (bf16)
    for (int f = 0; f < kF; ++f) {
      float v = ((float)xn_bf[xb + f] - mean2) * rs2 * l0g[f] + l0b[f];
      Abuf[r * kF + f] = (bf16_t)v;
    }
  }
  __syncthreads();

  gemm_block<128, 256, 128>(Abuf, w0, Cbuf);
  __syncthreads();
  if (r < 128) {  // elu + ffn_ln1
    float s = 0.f, ss = 0.f;
    for (int f = 0; f < kFH; ++f) {
      float h = eluf(Cbuf[r * kFH + f] + b0[f]);
      Cbuf[r * kFH + f] = h; s += h; ss += h * h;
    }
    float mean = s / kFH, rs = rsqrtf(ss / kFH - mean * mean + 1e-3f);
    for (int f = 0; f < kFH; ++f)
      Abuf[r * kFH + f] = (bf16_t)((Cbuf[r * kFH + f] - mean) * rs * l1g[f] + l1b[f]);
  }
  __syncthreads();

  gemm_block<128, 128, 128>(Abuf, w1, Cbuf);
  __syncthreads();
  if (r < 128) {  // elu + ffn_ln2
    float s = 0.f, ss = 0.f;
    for (int f = 0; f < kFH; ++f) {
      float h = eluf(Cbuf[r * kFH + f] + b1[f]);
      Cbuf[r * kFH + f] = h; s += h; ss += h * h;
    }
    float mean = s / kFH, rs = rsqrtf(ss / kFH - mean * mean + 1e-3f);
    for (int f = 0; f < kFH; ++f)
      Abuf[r * kFH + f] = (bf16_t)((Cbuf[r * kFH + f] - mean) * rs * l2g[f] + l2b[f]);
  }
  __syncthreads();

  gemm_block<128, 128, 128>(Abuf, w2, Cbuf);
  __syncthreads();
  if (r < 128) {  // x_dist + LSH argmax
    const int token = t0 + r;
    const size_t xdb = ((size_t)b * kN + token) * kDD;
    for (int f = 0; f < kDD; ++f) {
      float v = Cbuf[r * kDD + f] + b2[f];
      Cbuf[r * kDD + f] = v;
      xd_bf[xdb + f] = (bf16_t)v;
    }
    float mv[kNROT];
    for (int j = 0; j < kNROT; ++j) {
      float m = 0.f;
      for (int f = 0; f < kDD; ++f) m += Cbuf[r * kDD + f] * rot[f * kROTLD + j];
      mv[j] = m;
    }
    float best = -3.4e38f; int bi = 0;
    for (int j = 0; j < 2 * kNROT; ++j) {
      float v = (j < kNROT) ? mv[j] : -mv[j - kNROT];
      if (v > best) { best = v; bi = j; }
    }
    int bin = bi + (msk[b * kN + token] ? 0 : (kNBINS - 1));
    bin_idx[b * kN + token] = bin;
  }
}

// ---------------- k2: stable counting sort -> permutation (== argsort stable) ----------------
__global__ void k2a_hist(const int* __restrict__ bin_idx, int* __restrict__ counts) {
  __shared__ int lc[kNBKEY];
  const int b = blockIdx.x / kNCH, ch = blockIdx.x % kNCH;
  for (int i = threadIdx.x; i < kNBKEY; i += 256) lc[i] = 0;
  __syncthreads();
  const int bin = bin_idx[b * kN + ch * kBS + threadIdx.x];
  atomicAdd(&lc[bin], 1);
  __syncthreads();
  for (int i = threadIdx.x; i < kNBKEY; i += 256)
    counts[(b * kNCH + ch) * kNBKEY + i] = lc[i];
}

__global__ void k2b_scan(const int* __restrict__ counts, int* __restrict__ offsets) {
  __shared__ int binTot[kNBKEY];
  const int b = blockIdx.x, bin = threadIdx.x;
  if (bin < kNBKEY) {
    int run = 0;
    for (int c = 0; c < kNCH; ++c) {
      offsets[(b * kNCH + c) * kNBKEY + bin] = run;
      run += counts[(b * kNCH + c) * kNBKEY + bin];
    }
    binTot[bin] = run;
  }
  __syncthreads();
  if (threadIdx.x == 0) {
    int run = 0;
    for (int i = 0; i < kNBKEY; ++i) { int v = binTot[i]; binTot[i] = run; run += v; }
  }
  __syncthreads();
  if (bin < kNBKEY) {
    int s = binTot[bin];
    for (int c = 0; c < kNCH; ++c) offsets[(b * kNCH + c) * kNBKEY + bin] += s;
  }
}

__global__ void k2c_scatter(const int* __restrict__ bin_idx, const int* __restrict__ offsets,
                            int* __restrict__ perm) {
  __shared__ int lb[kBS];
  const int b = blockIdx.x / kNCH, ch = blockIdx.x % kNCH;
  const int t = ch * kBS + threadIdx.x;
  const int bin = bin_idx[b * kN + t];
  lb[threadIdx.x] = bin;
  __syncthreads();
  int rank = 0;
  for (int j = 0; j < (int)threadIdx.x; ++j) rank += (lb[j] == bin);
  const int dest = offsets[(b * kNCH + ch) * kNBKEY + bin] + rank;
  perm[b * kN + dest] = t;   // sorted position -> original token (bins_split)
}

// ---------------- k3: per-chunk pairwise Gaussian kernel (Gram via WMMA) ----------------
__global__ __launch_bounds__(256) void k3_pairwise(
    const bf16_t* __restrict__ xd_bf, const unsigned char* __restrict__ msk,
    const int* __restrict__ perm, bf16_t* __restrict__ dm, float* __restrict__ normw) {
  extern __shared__ char smem[];
  bf16_t* XM  = (bf16_t*)smem;                    // 256x128 bf16 (64KB)
  float* sq   = (float*)(smem + kBS * kDD * 2);   // 256
  float* mskb = sq + kBS;                         // 256
  float* deg  = mskb + kBS;                       // 256
  const int b = blockIdx.x / kNCH, ch = blockIdx.x % kNCH;
  const int r = threadIdx.x;
  {
    const int token = perm[b * kN + ch * kBS + r];
    const float m = msk[b * kN + token] ? 1.f : 0.f;
    const bf16_t* src = xd_bf + ((size_t)b * kN + token) * kDD;
    float s = 0.f;
#if ASYNC_LDS
    // async gather of this row (16B granules) directly into LDS, then scale in place
    for (int f = 0; f < kDD; f += 8) async_copy_b128(src + f, XM + r * kDD + f);
    async_wait0();
    for (int f = 0; f < kDD; ++f) {
      float v = (float)XM[r * kDD + f] * m;
      XM[r * kDD + f] = (bf16_t)v;
      s += v * v;
    }
#else
    for (int f = 0; f < kDD; ++f) {
      float v = (float)src[f] * m;
      XM[r * kDD + f] = (bf16_t)v;
      s += v * v;
    }
#endif
    sq[r] = s; mskb[r] = m; deg[r] = 0.f;
  }
  __syncthreads();
  const int wave = threadIdx.x >> 5, lane = threadIdx.x & 31;
  const size_t dmb = (size_t)blockIdx.x * kBS * kBS;
  for (int t = wave; t < 256; t += 8) {           // 16x16 tile grid over 256x256
    const int ti = t >> 4, tj = t & 15;
    v8f acc = {};
#pragma unroll
    for (int k0 = 0; k0 < kDD; k0 += 32) {
      v16bf a = load_a_frag(XM + ti * 16 * kDD + k0, kDD);
      v16bf bb = load_bt_frag(XM + tj * 16 * kDD + k0, kDD);  // B = XM^T
      acc = wmma_bf16(a, bb, acc);
    }
    const int col = tj * 16 + (lane & 15);
    const int rb  = ti * 16 + ((lane >> 4) << 3);
#pragma unroll
    for (int q = 0; q < 8; ++q) {
      const int row = rb + q;
      float d2 = sq[row] - 2.f * acc[q] + sq[col];
      float dv = expf(-0.1f * sqrtf(fmaxf(d2, 1e-6f)));
      dv = fminf(fmaxf(dv, 0.f), 1.f) * mskb[row] * mskb[col];
      dm[dmb + (size_t)row * kBS + col] = (bf16_t)dv;
      atomicAdd(&deg[row], dv);                   // dm >= 0 so |dm| == dm
    }
  }
  __syncthreads();
  {
    float dg = fminf(deg[r], 1000.f);
    normw[blockIdx.x * kBS + r] = rsqrtf(dg + 1e-6f) * mskb[r];
  }
}

// ---------------- k4: token-level GEMMs: Y=(xg@theta)*norm, f_het=xg@wh, gate ----------------
__global__ __launch_bounds__(256) void k4_token_gemms(
    const bf16_t* __restrict__ xn_bf, const unsigned char* __restrict__ msk,
    const int* __restrict__ perm, const float* __restrict__ normw,
    const bf16_t* __restrict__ theta, const bf16_t* __restrict__ wh,
    const bf16_t* __restrict__ wt, const float* __restrict__ bt,
    bf16_t* __restrict__ Yw, bf16_t* __restrict__ fhet, bf16_t* __restrict__ gate) {
  extern __shared__ char smem[];
  bf16_t* XG = (bf16_t*)smem;                     // 256x256 bf16 (128KB)
  float* nrm = (float*)(smem + kBS * kF * 2);     // 256
  const int b = blockIdx.x / kNCH, ch = blockIdx.x % kNCH;
  const int r = threadIdx.x;
  {
    const int token = perm[b * kN + ch * kBS + r];
    const float m = msk[b * kN + token] ? 1.f : 0.f;
    const bf16_t* src = xn_bf + ((size_t)b * kN + token) * kF;
#if ASYNC_LDS
    for (int f = 0; f < kF; f += 8) async_copy_b128(src + f, XG + r * kF + f);
    async_wait0();
    for (int f = 0; f < kF; ++f) XG[r * kF + f] = (bf16_t)((float)XG[r * kF + f] * m);
#else
    for (int f = 0; f < kF; ++f) XG[r * kF + f] = (bf16_t)((float)src[f] * m);
#endif
    nrm[r] = normw[blockIdx.x * kBS + r];
  }
  __syncthreads();
  const int wave = threadIdx.x >> 5, lane = threadIdx.x & 31;
  const size_t ob = (size_t)blockIdx.x * kBS * kOUT;
  for (int job = wave; job < 256 * 3; job += 8) {
    const int mtx = job / 256, t = job % 256;
    const int ti = t >> 4, tj = t & 15;
    const bf16_t* W = (mtx == 0) ? theta : ((mtx == 1) ? wh : wt);
    v8f acc = {};
#pragma unroll
    for (int k0 = 0; k0 < kF; k0 += 32) {
      v16bf a = load_a_frag(XG + ti * 16 * kF + k0, kF);
      v16bf bb = load_b_frag(W + k0 * kOUT + tj * 16, kOUT);
      acc = wmma_bf16(a, bb, acc);
    }
    const int col = tj * 16 + (lane & 15);
    const int rb  = ti * 16 + ((lane >> 4) << 3);
#pragma unroll
    for (int q = 0; q < 8; ++q) {
      const int row = rb + q;
      const size_t idx = ob + (size_t)row * kOUT + col;
      float v = acc[q];
      if (mtx == 0)      Yw[idx]   = (bf16_t)(v * nrm[row]);
      else if (mtx == 1) fhet[idx] = (bf16_t)v;
      else               gate[idx] = (bf16_t)(1.f / (1.f + expf(-(v + bt[col]))));
    }
  }
}

// ---------------- k5: f_hom = dm @ Y (WMMA) + gated elu + scatter to output ----------------
__global__ __launch_bounds__(256) void k5_fhom_out(
    const bf16_t* __restrict__ dm, const bf16_t* __restrict__ Yw,
    const bf16_t* __restrict__ fhet, const bf16_t* __restrict__ gate,
    const float* __restrict__ normw, const unsigned char* __restrict__ msk,
    const int* __restrict__ perm, float* __restrict__ out) {
  extern __shared__ char smem[];
  bf16_t* Ys  = (bf16_t*)smem;                    // 256x256 bf16 (128KB)
  float* nrm  = (float*)(smem + kBS * kOUT * 2);  // 256
  float* mskS = nrm + kBS;                        // 256
  int*   tokS = (int*)(mskS + kBS);               // 256
  const int b = blockIdx.x / kNCH, ch = blockIdx.x % kNCH;
  const size_t ob = (size_t)blockIdx.x * kBS * kOUT;
#if ASYNC_LDS
  // stage the whole Y panel into LDS via ASYNCcnt-tracked 16B copies
  for (int i = threadIdx.x * 8; i < kBS * kOUT; i += 256 * 8)
    async_copy_b128(Yw + ob + i, Ys + i);
  async_wait0();
#else
  for (int i = threadIdx.x; i < kBS * kOUT; i += 256) Ys[i] = Yw[ob + i];
#endif
  {
    const int r = threadIdx.x;
    const int token = perm[b * kN + ch * kBS + r];
    tokS[r] = token;
    mskS[r] = msk[b * kN + token] ? 1.f : 0.f;
    nrm[r]  = normw[blockIdx.x * kBS + r];
  }
  __syncthreads();
  const int wave = threadIdx.x >> 5, lane = threadIdx.x & 31;
  const size_t dmb = (size_t)blockIdx.x * kBS * kBS;
  for (int t = wave; t < 256; t += 8) {
    const int ti = t >> 4, tj = t & 15;
    v8f acc = {};
#pragma unroll
    for (int k0 = 0; k0 < kBS; k0 += 32) {
      v16bf a = load_a_frag(dm + dmb + (size_t)ti * 16 * kBS + k0, kBS);
      v16bf bb = load_b_frag(Ys + k0 * kOUT + tj * 16, kOUT);
      acc = wmma_bf16(a, bb, acc);
    }
    const int col = tj * 16 + (lane & 15);
    const int rb  = ti * 16 + ((lane >> 4) << 3);
#pragma unroll
    for (int q = 0; q < 8; ++q) {
      const int row = rb + q;
      const size_t idx = ob + (size_t)row * kOUT + col;
      float fhom = acc[q] * nrm[row];
      float g  = (float)gate[idx];
      float fh = (float)fhet[idx];
      float v  = g * fhom + (1.f - g) * fh;
      v = eluf(v) * mskS[row];
      out[((size_t)b * kN + tokS[row]) * kOUT + col] = v;  // perm is a bijection -> plain store
    }
  }
}

// ---------------- host ----------------
extern "C" void kernel_launch(void* const* d_in, const int* in_sizes, int n_in,
                              void* d_out, int out_size, void* d_ws, size_t ws_size,
                              hipStream_t stream) {
  (void)in_sizes; (void)n_in; (void)out_size; (void)ws_size;
  const float* x      = (const float*)d_in[0];
  const unsigned char* msk = (const unsigned char*)d_in[1];
  const float* ln1_g  = (const float*)d_in[2];
  const float* ln1_b  = (const float*)d_in[3];
  const float* l0g    = (const float*)d_in[4];
  const float* l0b    = (const float*)d_in[5];
  const float* w0f    = (const float*)d_in[6];
  const float* b0     = (const float*)d_in[7];
  const float* l1g    = (const float*)d_in[8];
  const float* l1b    = (const float*)d_in[9];
  const float* w1f    = (const float*)d_in[10];
  const float* b1     = (const float*)d_in[11];
  const float* l2g    = (const float*)d_in[12];
  const float* l2b    = (const float*)d_in[13];
  const float* w2f    = (const float*)d_in[14];
  const float* b2     = (const float*)d_in[15];
  const float* rot    = (const float*)d_in[16];
  const float* wtf    = (const float*)d_in[17];
  const float* bt     = (const float*)d_in[18];
  const float* whf    = (const float*)d_in[19];
  const float* thetaf = (const float*)d_in[20];
  float* out = (float*)d_out;
  char*  ws  = (char*)d_ws;

  // workspace carve-up (256B aligned)
  size_t off = 0;
  auto carve = [&](size_t bytes) { size_t o = off; off += (bytes + 255) & ~(size_t)255; return o; };
  bf16_t* xn_bf   = (bf16_t*)(ws + carve((size_t)kB * kN * kF * 2));
  bf16_t* xd_bf   = (bf16_t*)(ws + carve((size_t)kB * kN * kDD * 2));
  int*    bin_idx = (int*)   (ws + carve((size_t)kB * kN * 4));
  int*    perm    = (int*)   (ws + carve((size_t)kB * kN * 4));
  int*    counts  = (int*)   (ws + carve((size_t)kB * kNCH * kNBKEY * 4));
  int*    offsets = (int*)   (ws + carve((size_t)kB * kNCH * kNBKEY * 4));
  float*  normw   = (float*) (ws + carve((size_t)kB * kN * 4));
  bf16_t* dm      = (bf16_t*)(ws + carve((size_t)kB * kNCH * kBS * kBS * 2));
  bf16_t* Yw      = (bf16_t*)(ws + carve((size_t)kB * kN * kOUT * 2));
  bf16_t* fhet    = (bf16_t*)(ws + carve((size_t)kB * kN * kOUT * 2));
  bf16_t* gate    = (bf16_t*)(ws + carve((size_t)kB * kN * kOUT * 2));
  bf16_t* w0      = (bf16_t*)(ws + carve((size_t)kF * kFH * 2));
  bf16_t* w1      = (bf16_t*)(ws + carve((size_t)kFH * kFH * 2));
  bf16_t* w2      = (bf16_t*)(ws + carve((size_t)kFH * kDD * 2));
  bf16_t* theta   = (bf16_t*)(ws + carve((size_t)kF * kOUT * 2));
  bf16_t* wh      = (bf16_t*)(ws + carve((size_t)kF * kOUT * 2));
  bf16_t* wt      = (bf16_t*)(ws + carve((size_t)kF * kOUT * 2));

  // large dynamic-LDS kernels: raise the cap (no-op if unsupported)
  const int s1 = 128 * 256 * 2 + 128 * 128 * 4;                 // 128KB
  const int s3 = kBS * kDD * 2 + 3 * kBS * 4;                   // ~67KB
  const int s4 = kBS * kF * 2 + kBS * 4;                        // ~129KB
  const int s5 = kBS * kOUT * 2 + 3 * kBS * 4;                  // ~131KB
  (void)hipFuncSetAttribute((const void*)k1_ln_ffn_lsh,  hipFuncAttributeMaxDynamicSharedMemorySize, s1);
  (void)hipFuncSetAttribute((const void*)k3_pairwise,    hipFuncAttributeMaxDynamicSharedMemorySize, s3);
  (void)hipFuncSetAttribute((const void*)k4_token_gemms, hipFuncAttributeMaxDynamicSharedMemorySize, s4);
  (void)hipFuncSetAttribute((const void*)k5_fhom_out,    hipFuncAttributeMaxDynamicSharedMemorySize, s5);

  // k0: weight conversion fp32 -> bf16
  auto cvt = [&](const float* s, bf16_t* d, int n) {
    k0_cvt<<<(n + 255) / 256, 256, 0, stream>>>(s, d, n);
  };
  cvt(w0f, w0, kF * kFH);
  cvt(w1f, w1, kFH * kFH);
  cvt(w2f, w2, kFH * kDD);
  cvt(thetaf, theta, kF * kOUT);
  cvt(whf, wh, kF * kOUT);
  cvt(wtf, wt, kF * kOUT);

  // k1: LN + FFN + LSH
  k1_ln_ffn_lsh<<<kB * kNT1, 256, s1, stream>>>(
      x, msk, ln1_g, ln1_b, l0g, l0b, w0, b0, l1g, l1b, w1, b1,
      l2g, l2b, w2, b2, rot, xn_bf, xd_bf, bin_idx);

  // k2: stable counting sort -> perm (== argsort(bin_idx, stable=True))
  k2a_hist<<<kB * kNCH, 256, 0, stream>>>(bin_idx, counts);
  k2b_scan<<<kB, 128, 0, stream>>>(counts, offsets);
  k2c_scatter<<<kB * kNCH, 256, 0, stream>>>(bin_idx, offsets, perm);

  // k3: per-chunk pairwise Gaussian kernel + degree norm
  k3_pairwise<<<kB * kNCH, 256, s3, stream>>>(xd_bf, msk, perm, dm, normw);

  // k4: token-level GEMMs (theta / wh / wt)
  k4_token_gemms<<<kB * kNCH, 256, s4, stream>>>(
      xn_bf, msk, perm, normw, theta, wh, wt, bt, Yw, fhet, gate);

  // k5: dm @ Y, gated elu, scatter to output
  k5_fhom_out<<<kB * kNCH, 256, s5, stream>>>(
      dm, Yw, fhet, gate, normw, msk, perm, out);
}
